// HungarianMatcherDynamicK_8237747273838
// MI455X (gfx1250) — matrix operand order repaired and checked by reading, not compiled, not run on hardware
//
#include <hip/hip_runtime.h>
#include <hip/hip_bf16.h>
#include <stdint.h>

typedef _Float16 v16h __attribute__((ext_vector_type(16)));
typedef _Float16 v8h  __attribute__((ext_vector_type(8)));
typedef float    v8f  __attribute__((ext_vector_type(8)));
typedef int      ai4  __attribute__((ext_vector_type(4)));

constexpr int   B = 16, Q = 4096, G = 256, C = 80, CPAD = 96;
constexpr float K_CLASS = 2.0f, K_BBOX = 5.0f, K_GIOU = 2.0f, K_T = 1.0f, K_R = 1.0f;
constexpr float ALPHA = 0.25f;
constexpr int   OTA_K = 5;

#define AS1 __attribute__((address_space(1)))
#define AS3 __attribute__((address_space(3)))

#if __has_builtin(__builtin_amdgcn_global_load_async_to_lds_b128) && \
    __has_builtin(__builtin_amdgcn_global_load_async_to_lds_b32)  && \
    __has_builtin(__builtin_amdgcn_s_wait_asynccnt)
#define USE_ASYNC_LDS 1
#else
#define USE_ASYNC_LDS 0
#endif

// ---------------------------------------------------------------------------
// K0: focal class cost table  F[b][q][c] = pos - neg  (fp16, padded to CPAD)
// ---------------------------------------------------------------------------
__global__ void k_focal(const float* __restrict__ logits, _Float16* __restrict__ F, int total) {
  int idx = blockIdx.x * blockDim.x + threadIdx.x;
  if (idx >= total) return;
  int c  = idx % CPAD;
  int qb = idx / CPAD;           // b*Q + q
  float out = 0.0f;
  if (c < C) {
    float x   = logits[(size_t)qb * C + c];
    float p   = 1.0f / (1.0f + __expf(-x));
    float neg = (1.0f - ALPHA) * p * p * (-__logf(1.0f - p + 1e-8f));
    float pos = ALPHA * (1.0f - p) * (1.0f - p) * (-__logf(p + 1e-8f));
    out = pos - neg;
  }
  F[idx] = (_Float16)out;
}

// ---------------------------------------------------------------------------
// K1: cost + iou matrices, g-major [B][G][Q].
// GT tile (shared by all 8 waves) staged global->LDS with ASYNC loads.
// Class cost via WMMA: Focal(16xCPAD) @ OneHot(CPADx16), K split in 3x32.
// One wave computes one 16(q) x 16(g) tile.
// ---------------------------------------------------------------------------
__global__ __launch_bounds__(256) void k_cost(
    const _Float16* __restrict__ F,
    const float* __restrict__ boxes, const float* __restrict__ poses,
    const int*   __restrict__ labels,
    const float* __restrict__ gtb, const float* __restrict__ gtt,
    const float* __restrict__ gtr,
    const float* __restrict__ img, const float* __restrict__ imgt,
    float* __restrict__ cost, float* __restrict__ iouO)
{
  __shared__ __align__(16) float sgb[16][4];   // gt boxes xyxy
  __shared__ __align__(16) float sit[16][4];   // image_size_xyxy_tgt rows
  __shared__ __align__(16) float st [16][4];   // gt_trans (padded)
  __shared__ __align__(16) float sr [16][4];   // gt_rot   (padded)
  __shared__ __align__(16) int   slab[16];     // labels

  const int b    = blockIdx.z;
  const int tid  = threadIdx.x;
  const int wave = tid >> 5;
  const int lane = tid & 31;
  const int q0   = blockIdx.x * 128 + wave * 16;
  const int g0   = blockIdx.y * 16;
  const int n    = lane & 15;     // A-matrix row M, B/C/D column N
  const int hi   = lane >> 4;     // half-wave selector

  // ---- stage the 16-gt tile into LDS (once per block) ----
#if USE_ASYNC_LDS
  if (tid < 16) {
    const size_t gi = (size_t)(b * G + g0 + tid);
    __builtin_amdgcn_global_load_async_to_lds_b128(
        (AS1 ai4*)(gtb + gi * 4), (AS3 ai4*)&sgb[tid][0], 0, 0);
    __builtin_amdgcn_global_load_async_to_lds_b128(
        (AS1 ai4*)(imgt + gi * 4), (AS3 ai4*)&sit[tid][0], 0, 0);
    #pragma unroll
    for (int k = 0; k < 3; ++k) {
      __builtin_amdgcn_global_load_async_to_lds_b32(
          (AS1 int*)(gtt + gi * 3 + k), (AS3 int*)&st[tid][k], 0, 0);
      __builtin_amdgcn_global_load_async_to_lds_b32(
          (AS1 int*)(gtr + gi * 3 + k), (AS3 int*)&sr[tid][k], 0, 0);
    }
    __builtin_amdgcn_global_load_async_to_lds_b32(
        (AS1 int*)(labels + gi), (AS3 int*)&slab[tid], 0, 0);
  }
  __builtin_amdgcn_s_wait_asynccnt(0);
  __syncthreads();
#else
  if (tid < 16) {
    const size_t gi = (size_t)(b * G + g0 + tid);
    #pragma unroll
    for (int k = 0; k < 4; ++k) {
      sgb[tid][k] = gtb[gi * 4 + k];
      sit[tid][k] = imgt[gi * 4 + k];
    }
    #pragma unroll
    for (int k = 0; k < 3; ++k) {
      st[tid][k] = gtt[gi * 3 + k];
      sr[tid][k] = gtr[gi * 3 + k];
    }
    slab[tid] = labels[gi];
  }
  __syncthreads();
#endif

  // A-matrix source row: F[b][q0+n][*]; each half-wave fragment chunk is a
  // contiguous, 16B-aligned 8-half span -> two b128 loads per K-chunk.
  const _Float16* fr = F + (size_t)(b * Q + q0 + n) * CPAD;
  __builtin_prefetch(fr, 0, 3);  // -> global_prefetch_b8

  const int g   = g0 + n;
  const int lbl = slab[n];

  v8f acc = {};
  #pragma unroll
  for (int ch = 0; ch < 3; ++ch) {
    const int kb = ch * 32;
    // 16-bit A 16x32 layout: lanes0-15 K=0..7 & 16..23, lanes16-31 K=8..15 & 24..31
    const v8h lo = *(const v8h*)(fr + kb + hi * 8);
    const v8h hh = *(const v8h*)(fr + kb + 16 + hi * 8);
    const v16h a = __builtin_shufflevector(lo, hh, 0, 1, 2, 3, 4, 5, 6, 7,
                                           8, 9, 10, 11, 12, 13, 14, 15);
    // 16-bit B 32x16 layout (matches A): lanes0-15 K=0..7 & 16..23, lanes16-31 K=8..15 & 24..31, N = lane%16
    v16h bm;
    #pragma unroll
    for (int v = 0; v < 8; ++v) {
      const int k0 = kb + hi * 8 + 2 * v + (v >= 4 ? 8 : 0);
      bm[2 * v]     = (lbl == k0)     ? (_Float16)1.0f : (_Float16)0.0f;
      bm[2 * v + 1] = (lbl == k0 + 1) ? (_Float16)1.0f : (_Float16)0.0f;
    }
    acc = __builtin_amdgcn_wmma_f32_16x16x32_f16(false, a, false, bm,
                                                 (short)0, acc, false, false);
  }

  // ---- per-lane geometry: this lane owns column g, rows q0+hi*8 .. +7 ----
  const float4 gb = *(const float4*)&sgb[n][0];
  const float4 it = *(const float4*)&sit[n][0];
  const float t0 = st[n][0], t1 = st[n][1], t2 = st[n][2];
  const float r0 = sr[n][0], r1 = sr[n][1], r2 = sr[n][2];
  const float4 im = *(const float4*)(img + (size_t)b * 4);
  const float areaG = (gb.z - gb.x) * (gb.w - gb.y);
  const float gcx = 0.5f * (gb.x + gb.z), gcy = 0.5f * (gb.y + gb.w);
  const float gw = gb.z - gb.x, gh = gb.w - gb.y;
  const float gnx = gb.x / it.x, gny = gb.y / it.y;
  const float gnz = gb.z / it.z, gnw = gb.w / it.w;

  float cv[8], uv[8];
  #pragma unroll
  for (int i = 0; i < 8; ++i) {
    const int q = q0 + hi * 8 + i;
    const float4 p  = *(const float4*)(boxes + (size_t)(b * Q + q) * 4);
    const float* pp = poses + (size_t)(b * Q + q) * 6;
    float area1 = (p.z - p.x) * (p.w - p.y);
    float iw = fmaxf(fminf(p.z, gb.z) - fmaxf(p.x, gb.x), 0.0f);
    float ih = fmaxf(fminf(p.w, gb.w) - fmaxf(p.y, gb.y), 0.0f);
    float inter = iw * ih;
    float uni = area1 + areaG - inter;
    float iouv = inter / uni;
    float ew = fmaxf(fmaxf(p.z, gb.z) - fminf(p.x, gb.x), 0.0f);
    float eh = fmaxf(fmaxf(p.w, gb.w) - fminf(p.y, gb.y), 0.0f);
    float areaE = ew * eh;
    float giou = iouv - (areaE - uni) / areaE;
    float cb = fabsf(p.x / im.x - gnx) + fabsf(p.y / im.y - gny)
             + fabsf(p.z / im.z - gnz) + fabsf(p.w / im.w - gnw);
    float ct = fabsf(pp[0] - t0) + fabsf(pp[1] - t1) + fabsf(pp[2] - t2);
    float cr = fabsf(pp[3] - r0) + fabsf(pp[4] - r1) + fabsf(pp[5] - r2);
    float ax = 0.5f * (p.x + p.z), ay = 0.5f * (p.y + p.w);
    bool inb = (ax > gb.x) && (ax < gb.z) && (ay > gb.y) && (ay < gb.w);
    bool inc = (ax > gcx - 2.5f * gw) && (ax < gcx + 2.5f * gw) &&
               (ay > gcy - 2.5f * gh) && (ay < gcy + 2.5f * gh);
    cv[i] = K_BBOX * cb + K_CLASS * acc[i] - K_GIOU * giou
          + ((inb && inc) ? 0.0f : 100.0f) + K_T * ct + K_R * cr;
    uv[i] = iouv;
  }
  const size_t base = (size_t)(b * G + g) * Q + q0 + hi * 8;
  *(float4*)(cost + base)     = make_float4(cv[0], cv[1], cv[2], cv[3]);
  *(float4*)(cost + base + 4) = make_float4(cv[4], cv[5], cv[6], cv[7]);
  *(float4*)(iouO + base)     = make_float4(uv[0], uv[1], uv[2], uv[3]);
  *(float4*)(iouO + base + 4) = make_float4(uv[4], uv[5], uv[6], uv[7]);
}

// ---------------------------------------------------------------------------
// K2: foreground mask per (b,q):  0 if fg else 10000
// ---------------------------------------------------------------------------
__global__ void k_fg(const float* __restrict__ boxes, const float* __restrict__ gtb,
                     float* __restrict__ fgmask, int total) {
  int idx = blockIdx.x * blockDim.x + threadIdx.x;
  if (idx >= total) return;
  int b = idx / Q;
  const float4 p = *(const float4*)(boxes + (size_t)idx * 4);
  float ax = 0.5f * (p.x + p.z), ay = 0.5f * (p.y + p.w);
  bool any = false;
  const float* gB = gtb + (size_t)b * G * 4;
  for (int g = 0; g < G && !any; ++g) {
    float4 gb = *(const float4*)(gB + (size_t)g * 4);
    bool inb = (ax > gb.x) && (ax < gb.z) && (ay > gb.y) && (ay < gb.w);
    float gcx = 0.5f * (gb.x + gb.z), gcy = 0.5f * (gb.y + gb.w);
    float gw = gb.z - gb.x, gh = gb.w - gb.y;
    bool inc = (ax > gcx - 2.5f * gw) && (ax < gcx + 2.5f * gw) &&
               (ay > gcy - 2.5f * gh) && (ay < gcy + 2.5f * gh);
    any = inb || inc;
  }
  fgmask[idx] = any ? 0.0f : 10000.0f;
}

// K2b: broadcast fg penalty into cost (g-major, float4 vectorized, coalesced)
__global__ void k_fgadd(float* __restrict__ cost, const float* __restrict__ fgmask, int total4) {
  int idx = blockIdx.x * blockDim.x + threadIdx.x;
  if (idx >= total4) return;
  size_t e = (size_t)idx * 4;
  int q = (int)(e % Q);
  int b = (int)(e / ((size_t)Q * G));
  float4 c = *(float4*)(cost + e);
  const float4 f = *(const float4*)(fgmask + (size_t)b * Q + q);
  c.x += f.x; c.y += f.y; c.z += f.z; c.w += f.w;
  *(float4*)(cost + e) = c;
}

// ---------------------------------------------------------------------------
// K3: dynamic k per gt column: clamp(int(sum(top5 iou)), 1, ..). Wave/column.
// ---------------------------------------------------------------------------
__global__ __launch_bounds__(256) void k_dk(const float* __restrict__ iou, int* __restrict__ dk) {
  __shared__ float sm[8][32 * OTA_K];
  const int wave = threadIdx.x >> 5, lane = threadIdx.x & 31;
  const int cidx = blockIdx.x * 8 + wave;          // = b*G + g
  const float* col = iou + (size_t)cidx * Q;
  float t[OTA_K];
  #pragma unroll
  for (int k = 0; k < OTA_K; ++k) t[k] = -1.0f;
  for (int q = lane; q < Q; q += 32) {
    float v = col[q];
    if (v > t[OTA_K - 1]) {
      t[OTA_K - 1] = v;
      #pragma unroll
      for (int k = OTA_K - 1; k > 0; --k)
        if (t[k] > t[k - 1]) { float tmp = t[k]; t[k] = t[k - 1]; t[k - 1] = tmp; }
    }
  }
  #pragma unroll
  for (int k = 0; k < OTA_K; ++k) sm[wave][lane * OTA_K + k] = t[k];
  __syncthreads();
  if (lane == 0) {
    float m[OTA_K];
    #pragma unroll
    for (int k = 0; k < OTA_K; ++k) m[k] = -1.0f;
    for (int i = 0; i < 32 * OTA_K; ++i) {
      float v = sm[wave][i];
      if (v > m[OTA_K - 1]) {
        m[OTA_K - 1] = v;
        #pragma unroll
        for (int k = OTA_K - 1; k > 0; --k)
          if (m[k] > m[k - 1]) { float tmp = m[k]; m[k] = m[k - 1]; m[k - 1] = tmp; }
      }
    }
    float s = 0.0f;
    #pragma unroll
    for (int k = 0; k < OTA_K; ++k) s += m[k];
    int d = (int)s;
    dk[cidx] = d < 1 ? 1 : d;
  }
}

// ---------------------------------------------------------------------------
// K4: row argmin over g (loop-invariant since penalties are row-uniform)
// ---------------------------------------------------------------------------
__global__ void k_rowargmin(const float* __restrict__ cost, int* __restrict__ rowArg, int total) {
  int idx = blockIdx.x * blockDim.x + threadIdx.x;
  if (idx >= total) return;
  int b = idx / Q, q = idx % Q;
  const float* base = cost + (size_t)b * G * Q + q;
  float bv = base[0]; int bi = 0;
  for (int g = 1; g < G; ++g) {
    float v = base[(size_t)g * Q];
    if (v < bv) { bv = v; bi = g; }
  }
  rowArg[idx] = bi;
}

// ---------------------------------------------------------------------------
// K5: dynamic-k matching, one workgroup per batch. Assignment matrix lives in
// LDS as a Q x G bitmask (128KB) -- exercises CDNA5's 320KB-per-WGP LDS.
// ---------------------------------------------------------------------------
__global__ __launch_bounds__(1024) void k_match(
    const float* __restrict__ cost,      // [B][G][Q]
    const int* __restrict__ dk,          // [B][G]
    const int* __restrict__ rowArg,      // [B][Q]
    float* __restrict__ out_sel, float* __restrict__ out_gti, float* __restrict__ out_mq)
{
  extern __shared__ uint8_t smem[];
  uint32_t* mrow   = (uint32_t*)smem;                    // Q*8 words: per-row g-bitmask
  float*    pen    = (float*)(mrow + Q * 8);             // Q
  uint32_t* rowcnt = (uint32_t*)(pen + Q);               // Q
  uint32_t* colcnt = rowcnt + Q;                         // G
  uint32_t* stale  = colcnt + G;                         // Q/32 bits
  float*    redV   = (float*)(stale + Q / 32);           // 1024
  int*      redI   = (int*)(redV + 1024);                // 1024
  uint32_t* flags  = (uint32_t*)(redI + 1024);           // 2

  const int b = blockIdx.x, tid = threadIdx.x;
  const int wave = tid >> 5, lane = tid & 31;
  const float* costB = cost + (size_t)b * G * Q;

  for (int i = tid; i < Q * 8; i += 1024) mrow[i] = 0u;
  for (int q = tid; q < Q; q += 1024) { pen[q] = 0.0f; rowcnt[q] = 0u; }
  if (tid < G) colcnt[tid] = 0u;
  if (tid < Q / 32) stale[tid] = 0u;
  if (tid < 2) flags[tid] = 0u;
  __syncthreads();

  // ---- initial selection: dk[g] smallest costs per column ----
  for (int j = 0; j < 8; ++j) {
    const int col = wave * 8 + j;
    const int dkv = dk[b * G + col];
    const float* cc = costB + (size_t)col * Q;
    for (int pass = 0; pass < OTA_K; ++pass) {
      float bv = 3.4e38f; int bi = -1;
      if (pass < dkv) {
        for (int q = lane; q < Q; q += 32) {
          if ((mrow[q * 8 + (col >> 5)] >> (col & 31)) & 1u) continue;
          float v = cc[q];
          if (v < bv || (v == bv && (unsigned)q < (unsigned)bi)) { bv = v; bi = q; }
        }
      }
      redV[tid] = bv; redI[tid] = bi;
      __syncthreads();
      if (lane == 0 && pass < dkv) {
        float mv = redV[wave * 32]; int mi = redI[wave * 32];
        for (int l = 1; l < 32; ++l) {
          float v = redV[wave * 32 + l]; int i2 = redI[wave * 32 + l];
          if (i2 >= 0 && (mi < 0 || v < mv || (v == mv && i2 < mi))) { mv = v; mi = i2; }
        }
        if (mi >= 0) {
          atomicOr(&mrow[mi * 8 + (col >> 5)], 1u << (col & 31));
          atomicAdd(&rowcnt[mi], 1u);
          colcnt[col] += 1u;
        }
      }
      __syncthreads();
    }
  }

  // ---- stale rows (matched to >1 gt) -> one-hot of row argmin ----
  for (int q = tid; q < Q; q += 1024)
    if (rowcnt[q] > 1u) atomicOr(&stale[q >> 5], 1u << (q & 31));
  __syncthreads();
  for (int q = tid; q < Q; q += 1024) {
    if ((stale[q >> 5] >> (q & 31)) & 1u) {
      for (int w = 0; w < 8; ++w) {
        uint32_t bits = mrow[q * 8 + w];
        mrow[q * 8 + w] = 0u;
        while (bits) { int g2 = (w << 5) + __ffs(bits) - 1; bits &= bits - 1u; atomicSub(&colcnt[g2], 1u); }
      }
      int tg = rowArg[b * Q + q];
      mrow[q * 8 + (tg >> 5)] = 1u << (tg & 31);
      rowcnt[q] = 1u;
      atomicAdd(&colcnt[tg], 1u);
    }
  }
  __syncthreads();

  // ---- sequential fix-up loop (early exit when all gts matched) ----
  for (int it = 0; it < G; ++it) {
    if (tid == 0) { flags[0] = 0u; flags[1] = 0u; }
    __syncthreads();
    if (tid < G && colcnt[tid] == 0u) atomicOr(&flags[0], 1u);
    __syncthreads();
    if (!flags[0]) break;                       // "need" false: state frozen
    for (int q = tid; q < Q; q += 1024)         // accumulate row penalty
      if (rowcnt[q] > 0u) pen[q] += 100000.0f;
    __syncthreads();
    for (int j = 0; j < 8; ++j) {
      const int col = wave * 8 + j;
      const bool um = (colcnt[col] == 0u);
      const float* cc = costB + (size_t)col * Q;
      float bv = 3.4e38f; int bi = -1;
      if (um) {
        for (int q = lane; q < Q; q += 32) {
          float v = cc[q] + pen[q];
          if (v < bv || (v == bv && (unsigned)q < (unsigned)bi)) { bv = v; bi = q; }
        }
      }
      redV[tid] = bv; redI[tid] = bi;
      __syncthreads();
      if (lane == 0 && um) {
        float mv = redV[wave * 32]; int mi = redI[wave * 32];
        for (int l = 1; l < 32; ++l) {
          float v = redV[wave * 32 + l]; int i2 = redI[wave * 32 + l];
          if (i2 >= 0 && (mi < 0 || v < mv || (v == mv && i2 < mi))) { mv = v; mi = i2; }
        }
        if (mi >= 0) {
          atomicOr(&mrow[mi * 8 + (col >> 5)], 1u << (col & 31));
          atomicAdd(&rowcnt[mi], 1u);
          colcnt[col] = 1u;
        }
      }
      __syncthreads();
    }
    for (int q = tid; q < Q; q += 1024)
      if (rowcnt[q] > 1u) atomicOr(&flags[1], 1u);
    __syncthreads();
    if (flags[1]) {                              // any_conf: reset stale rows
      for (int q = tid; q < Q; q += 1024) {
        if ((stale[q >> 5] >> (q & 31)) & 1u) {
          for (int w = 0; w < 8; ++w) {
            uint32_t bits = mrow[q * 8 + w];
            mrow[q * 8 + w] = 0u;
            while (bits) { int g2 = (w << 5) + __ffs(bits) - 1; bits &= bits - 1u; atomicSub(&colcnt[g2], 1u); }
          }
          int tg = rowArg[b * Q + q];
          mrow[q * 8 + (tg >> 5)] = 1u << (tg & 31);
          rowcnt[q] = 1u;
          atomicAdd(&colcnt[tg], 1u);
        }
      }
    }
    __syncthreads();
  }
  __syncthreads();

  // ---- outputs ----
  for (int q = tid; q < Q; q += 1024) {
    out_sel[b * Q + q] = (rowcnt[q] > 0u) ? 1.0f : 0.0f;
    int gi = 0;
    for (int w = 0; w < 8; ++w) {
      uint32_t bits = mrow[q * 8 + w];
      if (bits) { gi = (w << 5) + __ffs(bits) - 1; break; }
    }
    out_gti[b * Q + q] = (float)gi;
  }
  for (int j = 0; j < 8; ++j) {
    const int col = wave * 8 + j;
    const float* cc = costB + (size_t)col * Q;
    float bv = 3.4e38f; int bi = -1;
    for (int q = lane; q < Q; q += 32) {
      if ((mrow[q * 8 + (col >> 5)] >> (col & 31)) & 1u) {
        float v = cc[q] + pen[q];
        if (v < bv || (v == bv && (unsigned)q < (unsigned)bi)) { bv = v; bi = q; }
      }
    }
    redV[tid] = bv; redI[tid] = bi;
    __syncthreads();
    if (lane == 0) {
      float mv = redV[wave * 32]; int mi = redI[wave * 32];
      for (int l = 1; l < 32; ++l) {
        float v = redV[wave * 32 + l]; int i2 = redI[wave * 32 + l];
        if (i2 >= 0 && (mi < 0 || v < mv || (v == mv && i2 < mi))) { mv = v; mi = i2; }
      }
      out_mq[b * G + col] = (float)(mi < 0 ? 0 : mi);
    }
    __syncthreads();
  }
}

// ---------------------------------------------------------------------------
extern "C" void kernel_launch(void* const* d_in, const int* in_sizes, int n_in,
                              void* d_out, int out_size, void* d_ws, size_t ws_size,
                              hipStream_t stream) {
  (void)in_sizes; (void)n_in; (void)out_size; (void)ws_size;
  const float* logits = (const float*)d_in[0];
  const float* boxes  = (const float*)d_in[1];
  const float* poses  = (const float*)d_in[2];
  const int*   labels = (const int*)  d_in[3];
  const float* gtb    = (const float*)d_in[4];
  const float* gtt    = (const float*)d_in[5];
  const float* gtr    = (const float*)d_in[6];
  const float* img    = (const float*)d_in[7];
  const float* imgt   = (const float*)d_in[8];

  uint8_t* ws = (uint8_t*)d_ws;
  size_t off = 0;
  auto take = [&](size_t bytes) -> uint8_t* {
    uint8_t* p = ws + off;
    off = (off + bytes + 255) & ~(size_t)255;
    return p;
  };
  _Float16* F   = (_Float16*)take((size_t)B * Q * CPAD * 2);
  float* cost   = (float*)take((size_t)B * G * Q * 4);
  float* iou    = (float*)take((size_t)B * G * Q * 4);
  float* fgmask = (float*)take((size_t)B * Q * 4);
  int*   dk     = (int*)take((size_t)B * G * 4);
  int*   rowArg = (int*)take((size_t)B * Q * 4);

  k_focal<<<(B * Q * CPAD + 255) / 256, 256, 0, stream>>>(logits, F, B * Q * CPAD);
  k_cost<<<dim3(Q / 128, G / 16, B), 256, 0, stream>>>(
      F, boxes, poses, labels, gtb, gtt, gtr, img, imgt, cost, iou);
  k_fg<<<(B * Q + 255) / 256, 256, 0, stream>>>(boxes, gtb, fgmask, B * Q);
  k_fgadd<<<(B * G * Q / 4 + 255) / 256, 256, 0, stream>>>(cost, fgmask, B * G * Q / 4);
  k_dk<<<(B * G) / 8, 256, 0, stream>>>(iou, dk);
  k_rowargmin<<<(B * Q) / 256, 256, 0, stream>>>(cost, rowArg, B * Q);

  float* out = (float*)d_out;
  const size_t smem = (size_t)Q * 8 * 4   // mrow bitmask
                    + (size_t)Q * 4       // pen
                    + (size_t)Q * 4       // rowcnt
                    + (size_t)G * 4       // colcnt
                    + (size_t)(Q / 32) * 4// stale
                    + 1024 * 4 + 1024 * 4 // reductions
                    + 64;                 // flags
  k_match<<<B, 1024, smem, stream>>>(cost, dk, rowArg,
                                     out, out + (size_t)B * Q, out + (size_t)2 * B * Q);
}
